// ManhattanAttention3D_10960756539557
// MI455X (gfx1250) — matrix operand order repaired and checked by reading, not compiled, not run on hardware
//
#include <hip/hip_runtime.h>

// MI455X (gfx1250, wave32) implementation.
// Stage 1: fold (w_sr,b_sr,bq,bkv) into one 192x64 bf16 weight + f32 bias,
// Stage 2: projection GEMM with v_wmma_f32_16x16x32_bf16 (f32 accumulate),
// Stage 3: per-(b,channel) attention with exact v_wmma_f32_16x16x4_f32,
//          two-pass online softmax over d (softmax axis is d per the einsum).
//          256 threads / 8 waves, 8 d-planes per chunk (~80 KB LDS; gfx1250
//          allows 320 KB per workgroup), global_prefetch of the next chunk.
// Workspace layout: [0,24K) Wf bf16 | [32K) bias f32 | [64K) q | +201MB k | +201MB v
// (requires ws_size >= ~604 MiB)

typedef __attribute__((ext_vector_type(16))) __bf16 v16bf;
typedef __attribute__((ext_vector_type(8)))  float  v8f;
typedef __attribute__((ext_vector_type(4)))  float  v4f;
typedef __attribute__((ext_vector_type(2)))  float  v2f;

#define Bsz 8
#define Cch 64
#define Hh  32
#define Wwi 32
#define Dd  96
#define Ssp (Hh*Wwi*Dd)      /* 98304 positions per (b,channel) */
#define SCALE 0.25f          /* hd^-0.5, hd=16 */

// ---------------------------------------------------------------------------
// Kernel 1: fold scale/shift into kv weights, convert to bf16, fold biases.
// Wf rows 0..63 = Wq ; rows 64..191 = Wkv * w_sr ; bias gets Wkv·b_sr folded.
// ---------------------------------------------------------------------------
__global__ void fold_kernel(const float* __restrict__ w_sr, const float* __restrict__ b_sr,
                            const float* __restrict__ Wq,   const float* __restrict__ bq,
                            const float* __restrict__ Wkv,  const float* __restrict__ bkv,
                            __bf16* __restrict__ Wf, float* __restrict__ biasf) {
  const int m = blockIdx.x;      // 0..191
  const int c = threadIdx.x;     // 0..63
  float v;
  if (m < 64) v = Wq[m*64 + c];
  else        v = Wkv[(m-64)*64 + c] * w_sr[c];
  Wf[m*64 + c] = (__bf16)v;
  if (c == 0) {
    float bias;
    if (m < 64) bias = bq[m];
    else {
      float acc = bkv[m - 64];
      for (int cc = 0; cc < 64; ++cc) acc += Wkv[(m-64)*64 + cc] * b_sr[cc];
      bias = acc;
    }
    biasf[m] = bias;
  }
}

// ---------------------------------------------------------------------------
// Kernel 2: projection GEMM.  D(192 x N) = Wf(192x64) * X(64 x N), N = B*H*W*D.
// One wave owns one 16-wide column tile; its X tile (B operand, bf16) is loaded
// once into registers (fully coalesced: 16 consecutive positions per half-wave)
// and reused across all 12 M-tiles (q:4, k:4, v:4).  K=64 -> 2 bf16 WMMAs/tile.
// ---------------------------------------------------------------------------
__global__ __launch_bounds__(128) void proj_kernel(const float*  __restrict__ x,
                                                   const __bf16* __restrict__ Wf,
                                                   const float*  __restrict__ biasf,
                                                   float* __restrict__ q,
                                                   float* __restrict__ k,
                                                   float* __restrict__ v) {
  const int lane = threadIdx.x & 31;
  const int wave = threadIdx.x >> 5;
  const int nl   = lane & 15;
  const int hi   = lane >> 4;                      // half-wave select
  const long long N0 = ((long long)blockIdx.x * 4 + wave) * 16;
  const int b  = (int)(N0 / Ssp);                  // Ssp % 16 == 0, no straddle
  const int s0 = (int)(N0 % Ssp);

  // B operand (bf16 32x16 per k-chunk): lane holds B[kbase+e][nl], kbase=ch*32+hi*16
  const float* xb = x + ((long long)b * Cch) * Ssp + s0 + nl;
  v16bf bmat[2];
#pragma unroll
  for (int ch = 0; ch < 2; ++ch) {
    const int kbase = ch * 32 + hi * 16;
#pragma unroll
    for (int e = 0; e < 16; ++e)
      bmat[ch][e] = (__bf16)xb[(long long)(kbase + e) * Ssp];
  }

  for (int mt = 0; mt < 12; ++mt) {
    const int m0 = mt * 16;
    // A operand (bf16 16x32 per k-chunk): lane holds row m0+nl,
    // elements 0..7 -> K = kb+e, elements 8..15 -> K = kb+8+e  (kb = ch*32+hi*16)
    const __bf16* arow = Wf + (m0 + nl) * 64;
    v16bf amat[2];
#pragma unroll
    for (int ch = 0; ch < 2; ++ch) {
      const int kb = ch * 32 + hi * 16;
#pragma unroll
      for (int e = 0; e < 8; ++e) {
        amat[ch][e]     = arow[kb + e];
        amat[ch][e + 8] = arow[kb + 8 + e];
      }
    }
    v8f acc;
#pragma unroll
    for (int r = 0; r < 8; ++r) acc[r] = biasf[m0 + r + hi*8];   // C/D: row = r + hi*8
    acc = __builtin_amdgcn_wmma_f32_16x16x32_bf16(false, amat[0], false, bmat[0],
                                                  (short)0, acc, false, false);
    acc = __builtin_amdgcn_wmma_f32_16x16x32_bf16(false, amat[1], false, bmat[1],
                                                  (short)0, acc, false, false);

    float* dst; int row;
    if (mt < 4)      { dst = q; row = m0; }
    else if (mt < 8) { dst = k; row = m0 - 64; }
    else             { dst = v; row = m0 - 128; }
    float* op = dst + ((long long)(b * Cch + row + hi*8)) * Ssp + s0 + nl;
#pragma unroll
    for (int r = 0; r < 8; ++r) op[(long long)r * Ssp] = acc[r];
  }
}

// ---------------------------------------------------------------------------
// Kernel 3: attention for one (b, channel) slice.  256 threads = 8 waves.
// d is processed in chunks of 8 planes; wave w owns plane d0+w.
// Pass 1: S = Q Kᵀ (contract over w, 32x32 per d) via f32 WMMA, online (m,Z)
//         over d kept in registers, cross-wave combined through LDS.
// Pass 2: recompute S, P = exp(S-m)/Z, O = P V via f32 WMMA, coalesced store.
// LDS planes padded to stride 33 -> conflict-free WMMA operand gathers.
// Next d-chunk is prefetched (global_prefetch_b8) while staging the current.
// ---------------------------------------------------------------------------
#define NWAVE 8
#define DCH   8
#define NCHUNK (Dd / DCH)     /* 12 */

__global__ __launch_bounds__(256) void attn_kernel(const float* __restrict__ qg,
                                                   const float* __restrict__ kg,
                                                   const float* __restrict__ vg,
                                                   const float* __restrict__ gamma,
                                                   const float* __restrict__ beta,
                                                   float* __restrict__ outg) {
  const int tid  = threadIdx.x;
  const int lane = tid & 31;
  const int wave = tid >> 5;          // 0..7 == local d-plane
  const int nl   = lane & 15;
  const int hi   = lane >> 4;

  const int bc   = blockIdx.x;        // 0..511
  const int b    = bc >> 6;
  const int ch   = bc & 63;
  const int head = ch >> 4;

  __shared__ float qs[NWAVE * 1056];  // Q planes / P planes / O planes (and m scratch)
  __shared__ float ks[NWAVE * 1056];  // K planes / V planes            (and Z scratch)
  __shared__ float decs[1024];
  __shared__ float mArr[1024];
  __shared__ float zArr[1024];        // holds 1/Z

  { // decay bias:  exp(-softplus(2|h-g|*gamma+beta)) == sigmoid(-(...))
    const float gam = gamma[head], bet = beta[head];
    for (int idx = tid; idx < 1024; idx += 256) {
      const int h = idx >> 5, g = idx & 31;
      const float dv = 2.0f * fabsf((float)(h - g)) * gam + bet;
      decs[idx] = 1.0f / (1.0f + __expf(dv));
    }
  }

  const long long base = ((long long)b * Cch + ch) * Ssp;
  const float* Q = qg + base;
  const float* K = kg + base;
  const float* V = vg + base;
  float*       O = outg + base;

  float* qp = qs + wave * 1056;       // this wave's plane
  float* kp = ks + wave * 1056;

  v8f mrun[2][2], zrun[2][2];
#pragma unroll
  for (int i = 0; i < 2; ++i)
#pragma unroll
    for (int j = 0; j < 2; ++j)
#pragma unroll
      for (int r = 0; r < 8; ++r) { mrun[i][j][r] = -__builtin_inff(); zrun[i][j][r] = 0.0f; }

  // ---------------- pass 1: softmax statistics over d ----------------
  for (int chunk = 0; chunk < NCHUNK; ++chunk) {
    const int d0 = chunk * DCH;
    __syncthreads();
    for (int pos = tid; pos < 1024; pos += 256) {     // coalesced 32B loads/position
      const int ofs = (pos >> 5) * 33 + (pos & 31);
      const float* qsrc = Q + pos * Dd + d0;
      const float* ksrc = K + pos * Dd + d0;
      v4f q0 = *(const v4f*)(qsrc);
      v4f q1 = *(const v4f*)(qsrc + 4);
      v4f k0 = *(const v4f*)(ksrc);
      v4f k1 = *(const v4f*)(ksrc + 4);
      __builtin_prefetch(qsrc + DCH, 0, 1);           // next chunk -> global_prefetch_b8
      __builtin_prefetch(ksrc + DCH, 0, 1);
#pragma unroll
      for (int dl = 0; dl < 4; ++dl) {
        qs[dl*1056 + ofs]     = q0[dl];
        qs[(dl+4)*1056 + ofs] = q1[dl];
        ks[dl*1056 + ofs]     = k0[dl];
        ks[(dl+4)*1056 + ofs] = k1[dl];
      }
    }
    __syncthreads();
#pragma unroll
    for (int i = 0; i < 2; ++i) {
#pragma unroll
      for (int j = 0; j < 2; ++j) {
        v8f acc = {0.f,0.f,0.f,0.f,0.f,0.f,0.f,0.f};
#pragma unroll
        for (int kk = 0; kk < 8; ++kk) {              // contract over w = 32
          const int w0 = kk * 4 + 2 * hi;             // f32 A/B layout: K = 2*hi + e
          v2f a, bm;
          a[0]  = qp[(i*16 + nl)*33 + w0];
          a[1]  = qp[(i*16 + nl)*33 + w0 + 1];
          bm[0] = kp[(j*16 + nl)*33 + w0];
          bm[1] = kp[(j*16 + nl)*33 + w0 + 1];
          acc = __builtin_amdgcn_wmma_f32_16x16x4_f32(false, a, false, bm,
                                                      (short)0, acc, false, false);
        }
#pragma unroll
        for (int r = 0; r < 8; ++r) {
          const int h = i*16 + r + hi*8, g = j*16 + nl;
          const float s  = acc[r] * SCALE + decs[h*32 + g];
          const float mo = mrun[i][j][r];
          const float mn = fmaxf(mo, s);
          zrun[i][j][r]  = zrun[i][j][r] * __expf(mo - mn) + __expf(s - mn);
          mrun[i][j][r]  = mn;
        }
      }
    }
  }

  // ---------------- cross-wave (m,Z) combine ----------------
  __syncthreads();
#pragma unroll
  for (int i = 0; i < 2; ++i)
#pragma unroll
    for (int j = 0; j < 2; ++j)
#pragma unroll
      for (int r = 0; r < 8; ++r) {
        const int hg = (i*16 + r + hi*8)*32 + j*16 + nl;
        qs[wave*1024 + hg] = mrun[i][j][r];
        ks[wave*1024 + hg] = zrun[i][j][r];
      }
  __syncthreads();
  for (int idx = tid; idx < 1024; idx += 256) {
    float mm = -__builtin_inff();
#pragma unroll
    for (int w = 0; w < NWAVE; ++w) mm = fmaxf(mm, qs[w*1024 + idx]);
    float zz = 0.0f;
#pragma unroll
    for (int w = 0; w < NWAVE; ++w) zz += ks[w*1024 + idx] * __expf(qs[w*1024 + idx] - mm);
    mArr[idx] = mm;
    zArr[idx] = 1.0f / zz;
  }

  // ---------------- pass 2: P = softmax(S), O = P V ----------------
  for (int chunk = 0; chunk < NCHUNK; ++chunk) {
    const int d0 = chunk * DCH;
    __syncthreads();
    for (int pos = tid; pos < 1024; pos += 256) {
      const int ofs = (pos >> 5) * 33 + (pos & 31);
      const float* qsrc = Q + pos * Dd + d0;
      const float* ksrc = K + pos * Dd + d0;
      v4f q0 = *(const v4f*)(qsrc);
      v4f q1 = *(const v4f*)(qsrc + 4);
      v4f k0 = *(const v4f*)(ksrc);
      v4f k1 = *(const v4f*)(ksrc + 4);
      __builtin_prefetch(qsrc + DCH, 0, 1);
      __builtin_prefetch(ksrc + DCH, 0, 1);
      __builtin_prefetch(V + pos * Dd + d0, 0, 1);    // V needed later this chunk
#pragma unroll
      for (int dl = 0; dl < 4; ++dl) {
        qs[dl*1056 + ofs]     = q0[dl];
        qs[(dl+4)*1056 + ofs] = q1[dl];
        ks[dl*1056 + ofs]     = k0[dl];
        ks[(dl+4)*1056 + ofs] = k1[dl];
      }
    }
    __syncthreads();
    // recompute S, form P (all 4 tiles before touching LDS again)
    v8f pt[2][2];
#pragma unroll
    for (int i = 0; i < 2; ++i) {
#pragma unroll
      for (int j = 0; j < 2; ++j) {
        v8f acc = {0.f,0.f,0.f,0.f,0.f,0.f,0.f,0.f};
#pragma unroll
        for (int kk = 0; kk < 8; ++kk) {
          const int w0 = kk * 4 + 2 * hi;
          v2f a, bm;
          a[0]  = qp[(i*16 + nl)*33 + w0];
          a[1]  = qp[(i*16 + nl)*33 + w0 + 1];
          bm[0] = kp[(j*16 + nl)*33 + w0];
          bm[1] = kp[(j*16 + nl)*33 + w0 + 1];
          acc = __builtin_amdgcn_wmma_f32_16x16x4_f32(false, a, false, bm,
                                                      (short)0, acc, false, false);
        }
#pragma unroll
        for (int r = 0; r < 8; ++r) {
          const int hg = (i*16 + r + hi*8)*32 + j*16 + nl;
          const float s = acc[r] * SCALE + decs[hg];
          acc[r] = __expf(s - mArr[hg]) * zArr[hg];
        }
        pt[i][j] = acc;
      }
    }
    // write P into this wave's q-plane as [h][g]
#pragma unroll
    for (int i = 0; i < 2; ++i)
#pragma unroll
      for (int j = 0; j < 2; ++j)
#pragma unroll
        for (int r = 0; r < 8; ++r)
          qp[(i*16 + r + hi*8)*33 + j*16 + nl] = pt[i][j][r];
    __syncthreads();                                   // K reads done; P visible
    for (int pos = tid; pos < 1024; pos += 256) {      // V chunk over K planes
      const int ofs = (pos >> 5) * 33 + (pos & 31);
      const float* vsrc = V + pos * Dd + d0;
      v4f v0 = *(const v4f*)(vsrc);
      v4f v1 = *(const v4f*)(vsrc + 4);
#pragma unroll
      for (int dl = 0; dl < 4; ++dl) {
        ks[dl*1056 + ofs]     = v0[dl];
        ks[(dl+4)*1056 + ofs] = v1[dl];
      }
    }
    __syncthreads();
    // O[h][j] = sum_g P[h][g] * V[g][j]
    v8f ot[2][2];
#pragma unroll
    for (int i = 0; i < 2; ++i) {
#pragma unroll
      for (int jt = 0; jt < 2; ++jt) {
        v8f acc = {0.f,0.f,0.f,0.f,0.f,0.f,0.f,0.f};
#pragma unroll
        for (int kk = 0; kk < 8; ++kk) {
          const int g0 = kk * 4 + 2 * hi;
          v2f a, bm;
          a[0]  = qp[(i*16 + nl)*33 + g0];             // P[h][g]
          a[1]  = qp[(i*16 + nl)*33 + g0 + 1];
          bm[0] = kp[g0*33       + jt*16 + nl];        // V[g][j]
          bm[1] = kp[(g0 + 1)*33 + jt*16 + nl];
          acc = __builtin_amdgcn_wmma_f32_16x16x4_f32(false, a, false, bm,
                                                      (short)0, acc, false, false);
        }
        ot[i][jt] = acc;
      }
    }
#pragma unroll
    for (int i = 0; i < 2; ++i)                        // O into q-plane as [h][j]
#pragma unroll
      for (int jt = 0; jt < 2; ++jt)
#pragma unroll
        for (int r = 0; r < 8; ++r)
          qp[(i*16 + r + hi*8)*33 + jt*16 + nl] = ot[i][jt][r];
    __syncthreads();
    for (int pos = tid; pos < 1024; pos += 256) {      // coalesced 32B stores
      const int ofs = (pos >> 5) * 33 + (pos & 31);
      v4f o0, o1;
#pragma unroll
      for (int dl = 0; dl < 4; ++dl) {
        o0[dl] = qs[dl*1056 + ofs];
        o1[dl] = qs[(dl+4)*1056 + ofs];
      }
      float* odst = O + pos * Dd + d0;
      *(v4f*)(odst)     = o0;
      *(v4f*)(odst + 4) = o1;
    }
  }
}

// ---------------------------------------------------------------------------
extern "C" void kernel_launch(void* const* d_in, const int* in_sizes, int n_in,
                              void* d_out, int out_size, void* d_ws, size_t ws_size,
                              hipStream_t stream) {
  const float* x     = (const float*)d_in[0];
  const float* w_sr  = (const float*)d_in[1];
  const float* b_sr  = (const float*)d_in[2];
  const float* Wq    = (const float*)d_in[3];
  const float* bq    = (const float*)d_in[4];
  const float* Wkv   = (const float*)d_in[5];
  const float* bkv   = (const float*)d_in[6];
  const float* gamma = (const float*)d_in[7];
  const float* beta  = (const float*)d_in[8];

  char*   ws    = (char*)d_ws;
  __bf16* Wf    = (__bf16*)ws;                      // 24 KiB
  float*  biasf = (float*)(ws + (32 << 10));        // 768 B
  float*  qbuf  = (float*)(ws + (64 << 10));
  const size_t plane = (size_t)Bsz * Cch * Ssp;     // 50,331,648 floats each
  float*  kbuf  = qbuf + plane;
  float*  vbuf  = kbuf + plane;

  fold_kernel<<<192, 64, 0, stream>>>(w_sr, b_sr, Wq, bq, Wkv, bkv, Wf, biasf);
  proj_kernel<<<12288, 128, 0, stream>>>(x, Wf, biasf, qbuf, kbuf, vbuf);
  attn_kernel<<<512, 256, 0, stream>>>(qbuf, kbuf, vbuf, gamma, beta, (float*)d_out);
}
